// Transformer_62947040690435
// MI455X (gfx1250) — matrix operand order repaired and checked by reading, not compiled, not run on hardware
//
#include <hip/hip_runtime.h>
#include <math.h>

typedef _Float16 half_t;
typedef __attribute__((ext_vector_type(16))) _Float16 v16h;
typedef __attribute__((ext_vector_type(8)))  _Float16 v8h;
typedef __attribute__((ext_vector_type(8)))  float    v8f;
typedef __attribute__((ext_vector_type(4)))  unsigned int u32x4;
typedef __attribute__((ext_vector_type(8)))  int      i32x8;
typedef __attribute__((ext_vector_type(4)))  int      i32x4;

#define NUM_BLOCKS 6
#define DM   1024
#define DFF  4096
#define NH   16
#define SEQL 33
#define BSZ  1024
#define DH   64
#define NTOK (BSZ * SEQL)   // 33792

#if defined(__has_builtin)
#if __has_builtin(__builtin_amdgcn_tensor_load_to_lds)
#define HAVE_TDM 1
#endif
#endif

// ---------------------------------------------------------------------------
// grid-stride f32 copy (x residual init; must not mutate d_in)
// ---------------------------------------------------------------------------
__global__ __launch_bounds__(256) void copy_f32(const float* __restrict__ src,
                                                float* __restrict__ dst, long n) {
  long i = (long)blockIdx.x * blockDim.x + threadIdx.x;
  long stride = (long)gridDim.x * blockDim.x;
  for (; i < n; i += stride) dst[i] = src[i];
}

// ---------------------------------------------------------------------------
// transpose + cast: src f32 [R][C] -> dst f16 [C][R]   (R,C multiples of 32)
// ---------------------------------------------------------------------------
__global__ __launch_bounds__(256) void transpose_cast(const float* __restrict__ src,
                                                      half_t* __restrict__ dst,
                                                      int R, int C) {
  __shared__ float tile[32][33];
  int c0 = blockIdx.x * 32, r0 = blockIdx.y * 32;
  int tx = threadIdx.x & 31, ty = threadIdx.x >> 5;   // 8 rows per pass
#pragma unroll
  for (int i = 0; i < 4; ++i) {
    int r = ty + i * 8;
    tile[r][tx] = src[(long)(r0 + r) * C + (c0 + tx)];
  }
  __syncthreads();
#pragma unroll
  for (int i = 0; i < 4; ++i) {
    int r = ty + i * 8;
    dst[(long)(c0 + r) * R + (r0 + tx)] = (half_t)tile[tx][r];
  }
}

// ---------------------------------------------------------------------------
// LayerNorm over D=1024, one block (256 threads) per token.
// outH != null -> write f16 ; outF != null -> write f32 (final LN to d_out)
// ---------------------------------------------------------------------------
__global__ __launch_bounds__(256) void ln_cast(const float* __restrict__ x,
                                               const float* __restrict__ sc,
                                               const float* __restrict__ sh,
                                               half_t* __restrict__ outH,
                                               float* __restrict__ outF) {
  __shared__ float red[256];
  __shared__ float smean, srstd;
  const int tid = threadIdx.x;
  const float* row = x + (long)blockIdx.x * DM;

  float v[4];
  float lsum = 0.f;
#pragma unroll
  for (int i = 0; i < 4; ++i) { v[i] = row[tid + i * 256]; lsum += v[i]; }
  red[tid] = lsum; __syncthreads();
  for (int off = 128; off; off >>= 1) {
    if (tid < off) red[tid] += red[tid + off];
    __syncthreads();
  }
  if (tid == 0) smean = red[0] * (1.0f / DM);
  __syncthreads();
  float mean = smean;
  float lsq = 0.f;
#pragma unroll
  for (int i = 0; i < 4; ++i) { float d = v[i] - mean; lsq += d * d; }
  __syncthreads();
  red[tid] = lsq; __syncthreads();
  for (int off = 128; off; off >>= 1) {
    if (tid < off) red[tid] += red[tid + off];
    __syncthreads();
  }
  if (tid == 0) srstd = rsqrtf(red[0] * (1.0f / DM) + 1e-5f);
  __syncthreads();
  float rstd = srstd;
#pragma unroll
  for (int i = 0; i < 4; ++i) {
    int c = tid + i * 256;
    float y = sc[c] * ((v[i] - mean) * rstd) + sh[c];
    if (outH) outH[(long)blockIdx.x * DM + c] = (half_t)y;
    if (outF) outF[(long)blockIdx.x * DM + c] = y;
  }
}

#ifdef HAVE_TDM
// ---------------------------------------------------------------------------
// Tensor Data Mover: post one 2-D tile load (tile_dim0=32 K-elems x
// tile_dim1=128 rows, f16) into LDS. D# per CDNA5 ISA ch.8 (groups 0/1;
// groups 2/3 unused for 2D -> zeros). Issued by one wave; tracked by
// TENSORcnt (in-order per wave). 6-arg builtin form (clang-23 headers).
// ---------------------------------------------------------------------------
__device__ __forceinline__ void tdm_load_tile32(const half_t* gptr,
                                                unsigned lds_off,
                                                unsigned K) {
  unsigned long long ga = (unsigned long long)(uintptr_t)gptr;
  u32x4 g0;
  g0[0] = 1u;                                          // count=1 (valid), user mode
  g0[1] = lds_off;                                     // lds_addr [63:32]
  g0[2] = (unsigned)(ga & 0xffffffffu);                // global_addr lo
  g0[3] = (unsigned)((ga >> 32) & 0x1ffffffu)          // global_addr hi (57-bit)
          | (2u << 30);                                // type=2 ("image")
  i32x8 g1;
  g1[0] = (int)(1u << 16);                             // data_size=1 -> 2 bytes
  g1[1] = (int)((K & 0xffffu) << 16);                  // tensor_dim0[15:0]
  g1[2] = (int)(((K >> 16) & 0xffffu)                  // tensor_dim0[31:16]
          | ((0x100000u & 0xffffu) << 16));            // tensor_dim1[15:0] (1<<20)
  g1[3] = (int)(((0x100000u >> 16) & 0xffffu)          // tensor_dim1[31:16]
          | (32u << 16));                              // tile_dim0 = 32
  g1[4] = 128;                                         // tile_dim1=128, tile_dim2=0
  g1[5] = (int)K;                                      // tensor_dim0_stride lo
  g1[6] = 0;                                           // stride0 hi | stride1 lo
  g1[7] = 0;
  i32x4 z4; z4[0] = 0; z4[1] = 0; z4[2] = 0; z4[3] = 0;
  i32x8 z8;
#pragma unroll
  for (int i = 0; i < 8; ++i) z8[i] = 0;
  __builtin_amdgcn_tensor_load_to_lds(g0, g1, z4, z4, z8, 0);
}
#endif

// ---------------------------------------------------------------------------
// WMMA GEMM:  C[M,N] = A[M,K](f16) * BT[N,K](f16)^T + bias[N]
//   block tile 128x128, 256 threads = 8 waves in 4(M) x 2(N),
//   wave tile 32x64 = 2x4 v_wmma_f32_16x16x32_f16 accumulators, BK = 32.
//   Tile staging: double-buffered TDM (tensor_load_to_lds + s_wait_tensorcnt),
//   falling back to synchronous vector staging if the builtin is absent.
// mode 0: outH = f16(C)        (QKV)
// mode 1: outH = f16(gelu(C))  (FFN up, exact erf GELU)
// mode 2: outF += C            (residual adds: proj / FFN down)
// M,N multiples of 128; K multiple of 32.
// ---------------------------------------------------------------------------
__global__ __launch_bounds__(256) void gemm_wmma(const half_t* __restrict__ A,
                                                 const half_t* __restrict__ BT,
                                                 const float* __restrict__ bias,
                                                 half_t* __restrict__ outH,
                                                 float* __restrict__ outF,
                                                 int M, int N, int K, int mode) {
#ifdef HAVE_TDM
  __shared__ half_t sA[2][128][32];
  __shared__ half_t sB[2][128][32];
#else
  __shared__ half_t sA[1][128][32];
  __shared__ half_t sB[1][128][32];
#endif

  const int tid = threadIdx.x;
  const int lane = tid & 31;
  const int wid = tid >> 5;
  const int wm = wid & 3;        // 0..3 -> 32-row slab
  const int wn = wid >> 2;       // 0..1 -> 64-col slab
  const int m0 = blockIdx.y * 128;
  const int n0 = blockIdx.x * 128;

  v8f acc[2][4];
#pragma unroll
  for (int tm = 0; tm < 2; ++tm)
#pragma unroll
    for (int tn = 0; tn < 4; ++tn) acc[tm][tn] = (v8f)0.f;

  const int lsel = (lane < 16) ? 0 : 8;    // A: K-half select (ISA 16-bit A layout)
  const int bks  = (lane < 16) ? 0 : 8;    // B: K-half select (ISA 16-bit B layout)
  const int arow0 = wm * 32 + (lane & 15);
  const int bcol0 = wn * 64 + (lane & 15);

#ifdef HAVE_TDM
  const unsigned ldsA[2] = { (unsigned)(uintptr_t)&sA[0][0][0],
                             (unsigned)(uintptr_t)&sA[1][0][0] };
  const unsigned ldsB[2] = { (unsigned)(uintptr_t)&sB[0][0][0],
                             (unsigned)(uintptr_t)&sB[1][0][0] };
  // prologue: post tile k=0 into buffer 0 (wave 0 only; EXEC ignored by TDM)
  if (wid == 0) {
    tdm_load_tile32(A  + (long)m0 * K, ldsA[0], (unsigned)K);
    tdm_load_tile32(BT + (long)n0 * K, ldsB[0], (unsigned)K);
  }
  int buf = 0;
  for (int k0 = 0; k0 < K; k0 += 32) {
    if (wid == 0) {
      if (k0 + 32 < K) {   // post next pair into the other buffer, then retire current
        tdm_load_tile32(A  + (long)m0 * K + (k0 + 32), ldsA[buf ^ 1], (unsigned)K);
        tdm_load_tile32(BT + (long)n0 * K + (k0 + 32), ldsB[buf ^ 1], (unsigned)K);
        __builtin_amdgcn_s_wait_tensorcnt(2);   // TDM in-order: current pair done
      } else {
        __builtin_amdgcn_s_wait_tensorcnt(0);
      }
    }
    __syncthreads();
#else
  const int srow = tid >> 1;
  const int scol = (tid & 1) * 16;
  int buf = 0;
  for (int k0 = 0; k0 < K; k0 += 32) {
    if (k0 + 32 < K) {
      __builtin_prefetch(A  + (long)(m0 + srow) * K + k0 + 32 + scol, 0, 1);
      __builtin_prefetch(BT + (long)(n0 + srow) * K + k0 + 32 + scol, 0, 1);
    }
    *(v16h*)&sA[0][srow][scol] = *(const v16h*)(A  + (long)(m0 + srow) * K + k0 + scol);
    *(v16h*)&sB[0][srow][scol] = *(const v16h*)(BT + (long)(n0 + srow) * K + k0 + scol);
    __syncthreads();
#endif

    v16h afrag[2];
#pragma unroll
    for (int tm = 0; tm < 2; ++tm) {
      union { v16h v; v8h h[2]; } au;
      au.h[0] = *(const v8h*)&sA[buf][arow0 + tm * 16][lsel];        // K 0..7 / 8..15
      au.h[1] = *(const v8h*)&sA[buf][arow0 + tm * 16][16 + lsel];   // K 16..23 / 24..31
      afrag[tm] = au.v;
    }
#pragma unroll
    for (int tn = 0; tn < 4; ++tn) {
      union { v16h v; v8h h[2]; } bu;
      bu.h[0] = *(const v8h*)&sB[buf][bcol0 + tn * 16][bks];
      bu.h[1] = *(const v8h*)&sB[buf][bcol0 + tn * 16][16 + bks];
      v16h bfrag = bu.v;
#pragma unroll
      for (int tm = 0; tm < 2; ++tm) {
        acc[tm][tn] = __builtin_amdgcn_wmma_f32_16x16x32_f16(
            false, afrag[tm], false, bfrag, (short)0, acc[tm][tn], false, false);
      }
    }
    __syncthreads();
#ifdef HAVE_TDM
    buf ^= 1;
#endif
  }

  // epilogue — C/D layout: lane<16: N=lane, M=reg; lane>=16: N=lane-16, M=reg+8
  const int roff = (lane >= 16) ? 8 : 0;
  const int ncol = lane & 15;
#pragma unroll
  for (int tn = 0; tn < 4; ++tn) {
    const int n = n0 + wn * 64 + tn * 16 + ncol;
    const float bv = bias[n];
#pragma unroll
    for (int tm = 0; tm < 2; ++tm) {
#pragma unroll
      for (int r = 0; r < 8; ++r) {
        const int m = m0 + wm * 32 + tm * 16 + r + roff;
        float val = acc[tm][tn][r] + bv;
        if (mode == 1) val = 0.5f * val * (1.0f + erff(val * 0.70710678118654752f));
        if (mode == 2) {
          outF[(long)m * N + n] += val;
        } else {
          outH[(long)m * N + n] = (half_t)val;
        }
      }
    }
  }
}

// ---------------------------------------------------------------------------
// Attention: one block per (head, batch). S=33, Dh=64.
// qkv f16 [NTOK][3*DM] -> o f16 [NTOK][DM]
// scores = (q.k^T)/8 * m - (1-m)*1e10 ; softmax ; o = attn @ v
// ---------------------------------------------------------------------------
__global__ __launch_bounds__(256) void attn_kernel(const half_t* __restrict__ qkv,
                                                   const int* __restrict__ mask,
                                                   half_t* __restrict__ o) {
  __shared__ half_t sq[SEQL][DH];
  __shared__ half_t sk[SEQL][DH];
  __shared__ half_t sv[SEQL][DH];
  __shared__ float  scs[SEQL][SEQL + 1];

  const int h = blockIdx.x;
  const int b = blockIdx.y;
  const int tid = threadIdx.x;

  for (int idx = tid; idx < SEQL * DH; idx += 256) {
    int s = idx / DH, d = idx % DH;
    long base = (long)(b * SEQL + s) * (3 * DM) + h * DH + d;
    sq[s][d] = qkv[base];
    sk[s][d] = qkv[base + DM];
    sv[s][d] = qkv[base + 2 * DM];
  }
  __syncthreads();

  for (int e = tid; e < SEQL * SEQL; e += 256) {
    int i = e / SEQL, j = e % SEQL;
    float dot = 0.f;
#pragma unroll
    for (int d = 0; d < DH; ++d) dot += (float)sq[i][d] * (float)sk[j][d];
    float m = (float)mask[i * SEQL + j];
    scs[i][j] = dot * 0.125f * m - (1.0f - m) * 1e10f;
  }
  __syncthreads();

  if (tid < SEQL) {
    float mx = -3.4e38f;
    for (int j = 0; j < SEQL; ++j) mx = fmaxf(mx, scs[tid][j]);
    float sum = 0.f;
    for (int j = 0; j < SEQL; ++j) { float e = __expf(scs[tid][j] - mx); scs[tid][j] = e; sum += e; }
    float inv = 1.0f / sum;
    for (int j = 0; j < SEQL; ++j) scs[tid][j] *= inv;
  }
  __syncthreads();

  for (int e = tid; e < SEQL * DH; e += 256) {
    int i = e / DH, d = e % DH;
    float acc = 0.f;
    for (int j = 0; j < SEQL; ++j) acc += scs[i][j] * (float)sv[j][d];
    o[(long)(b * SEQL + i) * DM + h * DH + d] = (half_t)acc;
  }
}

// ---------------------------------------------------------------------------
// host orchestration
// ---------------------------------------------------------------------------
extern "C" void kernel_launch(void* const* d_in, const int* in_sizes, int n_in,
                              void* d_out, int out_size, void* d_ws, size_t ws_size,
                              hipStream_t stream) {
  (void)in_sizes; (void)n_in; (void)out_size; (void)ws_size;
  const float* x_in   = (const float*)d_in[0];
  const float* qkv_w  = (const float*)d_in[1];
  const float* qkv_b  = (const float*)d_in[2];
  const float* proj_w = (const float*)d_in[3];
  const float* proj_b = (const float*)d_in[4];
  const float* w1     = (const float*)d_in[5];
  const float* b1     = (const float*)d_in[6];
  const float* w2     = (const float*)d_in[7];
  const float* b2     = (const float*)d_in[8];
  const float* ln1_s  = (const float*)d_in[9];
  const float* ln1_b  = (const float*)d_in[10];
  const float* ln2_s  = (const float*)d_in[11];
  const float* ln2_b  = (const float*)d_in[12];
  const float* lnf_s  = (const float*)d_in[13];
  const float* lnf_b  = (const float*)d_in[14];
  const int*   mask0  = (const int*)d_in[15];
  const int*   maskr  = (const int*)d_in[16];

  char* p = (char*)d_ws;
  auto take = [&p](size_t bytes) {
    void* r = (void*)p;
    p += (bytes + 255) & ~(size_t)255;
    return r;
  };
  const long M = NTOK;
  float*  X     = (float*) take((size_t)M * DM * 4);
  half_t* H16   = (half_t*)take((size_t)M * DM * 2);
  half_t* QKV16 = (half_t*)take((size_t)M * 3 * DM * 2);
  half_t* O16   = (half_t*)take((size_t)M * DM * 2);
  half_t* U16   = (half_t*)take((size_t)M * DFF * 2);
  half_t* WqkvT = (half_t*)take((size_t)NUM_BLOCKS * 3 * DM * DM * 2);
  half_t* WprT  = (half_t*)take((size_t)NUM_BLOCKS * DM * DM * 2);
  half_t* W1T   = (half_t*)take((size_t)NUM_BLOCKS * DFF * DM * 2);
  half_t* W2T   = (half_t*)take((size_t)NUM_BLOCKS * DM * DFF * 2);

  // residual stream init (don't mutate d_in)
  copy_f32<<<4096, 256, 0, stream>>>(x_in, X, M * (long)DM);

  // one-time (per launch) weight transpose+cast to f16 [N][K]
  for (int l = 0; l < NUM_BLOCKS; ++l) {
    transpose_cast<<<dim3(3 * DM / 32, DM / 32), 256, 0, stream>>>(
        qkv_w + (long)l * DM * 3 * DM, WqkvT + (long)l * 3 * DM * DM, DM, 3 * DM);
    transpose_cast<<<dim3(DM / 32, DM / 32), 256, 0, stream>>>(
        proj_w + (long)l * DM * DM, WprT + (long)l * DM * DM, DM, DM);
    transpose_cast<<<dim3(DFF / 32, DM / 32), 256, 0, stream>>>(
        w1 + (long)l * DM * DFF, W1T + (long)l * DFF * DM, DM, DFF);
    transpose_cast<<<dim3(DM / 32, DFF / 32), 256, 0, stream>>>(
        w2 + (long)l * DFF * DM, W2T + (long)l * DM * DFF, DFF, DM);
  }

  const dim3 blk(256);
  for (int l = 0; l < NUM_BLOCKS; ++l) {
    // LN1 -> f16
    ln_cast<<<M, blk, 0, stream>>>(X, ln1_s + l * DM, ln1_b + l * DM, H16, nullptr);
    // QKV = h @ Wqkv + b          [M,3072]
    gemm_wmma<<<dim3(3 * DM / 128, M / 128), blk, 0, stream>>>(
        H16, WqkvT + (long)l * 3 * DM * DM, qkv_b + l * 3 * DM,
        QKV16, nullptr, M, 3 * DM, DM, 0);
    // attention
    attn_kernel<<<dim3(NH, BSZ), blk, 0, stream>>>(
        QKV16, (l == 0) ? mask0 : maskr, O16);
    // x += o @ Wproj + b
    gemm_wmma<<<dim3(DM / 128, M / 128), blk, 0, stream>>>(
        O16, WprT + (long)l * DM * DM, proj_b + l * DM,
        nullptr, X, M, DM, DM, 2);
    // LN2 -> f16
    ln_cast<<<M, blk, 0, stream>>>(X, ln2_s + l * DM, ln2_b + l * DM, H16, nullptr);
    // u = gelu(h @ W1 + b1)       [M,4096]
    gemm_wmma<<<dim3(DFF / 128, M / 128), blk, 0, stream>>>(
        H16, W1T + (long)l * DFF * DM, b1 + l * DFF,
        U16, nullptr, M, DFF, DM, 1);
    // x += u @ W2 + b2
    gemm_wmma<<<dim3(DM / 128, M / 128), blk, 0, stream>>>(
        U16, W2T + (long)l * DM * DFF, b2 + l * DM,
        nullptr, X, M, DM, DFF, 2);
  }
  // final LN -> f32 output
  ln_cast<<<M, blk, 0, stream>>>(X, lnf_s, lnf_b, nullptr, (float*)d_out);
}
